// MSAModel_41248865911236
// MI455X (gfx1250) — matrix operand order repaired and checked by reading, not compiled, not run on hardware
//
#include <hip/hip_runtime.h>
#include <stdint.h>

// ---------------------------------------------------------------------------
// Problem constants (from reference setup_inputs)
// ---------------------------------------------------------------------------
#define L_  18      // MSA layers
#define B_  16      // query batch
#define SQ_ 256     // query tokens
#define H_  2560    // hidden dim
#define R_  256     // router dim
#define D_  2048    // docs
#define C_  32      // chunks per doc
#define TOPK_ 16

typedef __attribute__((ext_vector_type(16))) __bf16 v16bf;
typedef __attribute__((ext_vector_type(8)))  float  v8f;
typedef __attribute__((ext_vector_type(4)))  unsigned u32x4;
typedef __attribute__((ext_vector_type(8)))  unsigned u32x8;

struct BFrag  { uint4 lo, hi; };        // 32 bytes == v16bf
struct U8Frag { unsigned w[8]; };       // 32 bytes == v16bf

// wait for outstanding async global->LDS ops to drop to <= n
#if __has_builtin(__builtin_amdgcn_s_wait_asynccnt)
#define WAIT_ASYNC(n) __builtin_amdgcn_s_wait_asynccnt(n)
#else
#define WAIT_ASYNC(n) asm volatile("s_wait_asynccnt %0" :: "i"(n) : "memory")
#endif
// wait for outstanding TDM (tensor DMA) ops to drop to <= n
#if __has_builtin(__builtin_amdgcn_s_wait_tensorcnt)
#define WAIT_TENSOR(n) __builtin_amdgcn_s_wait_tensorcnt(n)
#else
#define WAIT_TENSOR(n) asm volatile("s_wait_tensorcnt %0" :: "i"(n) : "memory")
#endif
#define RFL(x) ((unsigned)__builtin_amdgcn_readfirstlane((int)(x)))

// f32 -> bf16 (round to nearest even)
__device__ inline unsigned pk_bf16x2(float a, float b) {
  unsigned ua = __builtin_bit_cast(unsigned, a);
  unsigned ub = __builtin_bit_cast(unsigned, b);
  ua = (ua + 0x7FFFu + ((ua >> 16) & 1u)) >> 16;
  ub = (ub + 0x7FFFu + ((ub >> 16) & 1u)) >> 16;
  return (ua & 0xFFFFu) | (ub << 16);
}
__device__ inline unsigned short f2bf(float a) {
  unsigned ua = __builtin_bit_cast(unsigned, a);
  ua = (ua + 0x7FFFu + ((ua >> 16) & 1u)) >> 16;
  return (unsigned short)ua;
}
// order-preserving float <-> uint key (for exact atomic max on floats)
__device__ inline unsigned f2mono(float f) {
  unsigned u = __builtin_bit_cast(unsigned, f);
  return (u & 0x80000000u) ? ~u : (u | 0x80000000u);
}
__device__ inline float mono2f(unsigned k) {
  unsigned u = (k & 0x80000000u) ? (k & 0x7FFFFFFFu) : ~k;
  return __builtin_bit_cast(float, u);
}

// ---------------------------------------------------------------------------
// Kernel 0: init monotone-max keys to "-inf" (key 0)
// ---------------------------------------------------------------------------
__global__ void k0_init(unsigned* __restrict__ qkey) {
  qkey[(size_t)blockIdx.x * 256 + threadIdx.x] = 0u;
}

// ---------------------------------------------------------------------------
// Kernel 1: q_vec_raw[b,l,r] = max_s  sum_h q_hs[b,l,s,h] * w_q[l,r,h]
// grid = (2 Nblk, 2 Mblk, B*L), 256 threads = 8 waves.
// Block GEMM tile: 128(M=s) x 128(N=r), K = H = 2560, bf16 WMMA, f32 acc.
// LDS double buffered, Kstage = 64, f32->bf16 conversion on the fill path.
// ---------------------------------------------------------------------------
__global__ __launch_bounds__(256) void k1_project(const float* __restrict__ qhs,
                                                  const float* __restrict__ wq,
                                                  unsigned* __restrict__ qkey) {
  extern __shared__ __align__(16) char smem[];
  constexpr int SA = 80;    // bf16 elems per A row (64 + pad), 160B: 16B aligned
  constexpr int SB = 144;   // bf16 elems per B row (128 + pad), 288B: 16B aligned
  constexpr int KST = 64;
  unsigned short* sA = (unsigned short*)smem;                     // [2][128][SA]
  unsigned short* sB = (unsigned short*)(smem + 2 * 128 * SA * 2);// [2][64][SB]

  const int t = threadIdx.x, lane = t & 31, wid = t >> 5;
  const int nblk = blockIdx.x, mblk = blockIdx.y, bl = blockIdx.z;
  const int l = bl % L_;
  const int wm = wid >> 2;      // 0..1 : wave M offset (64 rows)
  const int wn = wid & 3;       // 0..3 : wave N offset (32 cols)
  const size_t abase = (size_t)bl * SQ_ * H_ + (size_t)(mblk * 128) * H_;
  const size_t wbase = (size_t)l * R_ * H_ + (size_t)(nblk * 128) * H_;
  const int rowg = t >> 3;            // 0..31
  const int kk   = (t & 7) * 8;       // 0..56

  float4 pa[4][2], pb[4][2];
  auto loadStage = [&](int kb) {
    #pragma unroll
    for (int p = 0; p < 4; ++p) {
      const float* ap = qhs + abase + (size_t)(p * 32 + rowg) * H_ + kb + kk;
      pa[p][0] = *(const float4*)ap;
      pa[p][1] = *(const float4*)(ap + 4);
      const float* bp = wq + wbase + (size_t)(p * 32 + rowg) * H_ + kb + kk;
      pb[p][0] = *(const float4*)bp;
      pb[p][1] = *(const float4*)(bp + 4);
    }
  };
  auto storeStage = [&](int buf) {
    unsigned short* A  = sA + buf * 128 * SA;
    unsigned short* Bm = sB + buf * 64 * SB;
    #pragma unroll
    for (int p = 0; p < 4; ++p) {
      uint4 w;
      w.x = pk_bf16x2(pa[p][0].x, pa[p][0].y);
      w.y = pk_bf16x2(pa[p][0].z, pa[p][0].w);
      w.z = pk_bf16x2(pa[p][1].x, pa[p][1].y);
      w.w = pk_bf16x2(pa[p][1].z, pa[p][1].w);
      *(uint4*)(A + (p * 32 + rowg) * SA + kk) = w;     // A: [m][k], row-major k
      // B transposed at store: sB[k][n] = w_q[l, n, k]
      float bv[8] = { pb[p][0].x, pb[p][0].y, pb[p][0].z, pb[p][0].w,
                      pb[p][1].x, pb[p][1].y, pb[p][1].z, pb[p][1].w };
      #pragma unroll
      for (int j = 0; j < 8; ++j)
        Bm[(kk + j) * SB + (p * 32 + rowg)] = f2bf(bv[j]);
    }
  };

  const v8f z = {0.f,0.f,0.f,0.f,0.f,0.f,0.f,0.f};
  v8f acc[4][2];
  #pragma unroll
  for (int i = 0; i < 4; ++i)
    #pragma unroll
    for (int j = 0; j < 2; ++j) acc[i][j] = z;

  loadStage(0);
  storeStage(0);
  __syncthreads();

  const int NST = H_ / KST;  // 40
  for (int s = 0; s < NST; ++s) {
    if (s + 1 < NST) loadStage((s + 1) * KST);
    const unsigned short* A  = sA + (s & 1) * 128 * SA;
    const unsigned short* Bm = sB + (s & 1) * 64 * SB;
    #pragma unroll
    for (int ks = 0; ks < 2; ++ks) {
      // preload ALL fragments of this k-substep, then a dense WMMA burst
      v16bf bfr[2], afr[4];
      #pragma unroll
      for (int j = 0; j < 2; ++j) {      // B fragment: lane holds column n=lane&15, k in table order
        const unsigned short* p = Bm + (ks * 32 + (lane >> 4) * 8) * SB + wn * 32 + j * 16 + (lane & 15);
        v16bf bt;
        #pragma unroll
        for (int e = 0; e < 16; ++e)
          bt[e] = __builtin_bit_cast(__bf16, p[((e & 7) + ((e >> 3) << 4)) * SB]);
        bfr[j] = bt;
      }
      #pragma unroll
      for (int i = 0; i < 4; ++i) {      // A fragment: lane half selects k sub-block
        const unsigned short* p =
            A + (wm * 64 + i * 16 + (lane & 15)) * SA + ks * 32 + (lane >> 4) * 8;
        BFrag f; f.lo = *(const uint4*)p; f.hi = *(const uint4*)(p + 16);
        afr[i] = __builtin_bit_cast(v16bf, f);
      }
      #pragma unroll
      for (int i = 0; i < 4; ++i)
        #pragma unroll
        for (int j = 0; j < 2; ++j)
          acc[i][j] = __builtin_amdgcn_wmma_f32_16x16x32_bf16(
              false, afr[i], false, bfr[j], (short)0, acc[i][j], false, false);
    }
    __syncthreads();
    if (s + 1 < NST) storeStage((s + 1) & 1);
    __syncthreads();
  }

  // max over M (token axis) -> atomic max into qkey[b,l,r]
  #pragma unroll
  for (int j = 0; j < 2; ++j) {
    float m = -3.402823466e38f;
    #pragma unroll
    for (int i = 0; i < 4; ++i)
      #pragma unroll
      for (int r = 0; r < 8; ++r) m = fmaxf(m, acc[i][j][r]);
    m = fmaxf(m, __shfl_xor(m, 16, 32));  // combine lane halves (M and M+8)
    if (lane < 16) {
      const int r = nblk * 128 + wn * 32 + j * 16 + lane;
      atomicMax(&qkey[(size_t)bl * R_ + r], f2mono(m));
    }
  }
}

// ---------------------------------------------------------------------------
// Kernel 2: L2-normalize q_vec per (b,l); emit bf16 pre-transposed [l][r][b]
// (exactly the WMMA B-operand layout for kernel 3). 288 blocks x 1 wave.
// ---------------------------------------------------------------------------
__global__ __launch_bounds__(32) void k2_norm(const unsigned* __restrict__ qkey,
                                              unsigned short* __restrict__ qb) {
  const int t = threadIdx.x, bl = blockIdx.x;
  const int b = bl / L_, l = bl % L_;
  float v[8]; float ss = 0.f;
  #pragma unroll
  for (int j = 0; j < 8; ++j) {
    v[j] = mono2f(qkey[(size_t)bl * R_ + t + j * 32]);
    ss += v[j] * v[j];
  }
  #pragma unroll
  for (int o = 16; o > 0; o >>= 1) ss += __shfl_xor(ss, o, 32);
  const float inv = 1.f / fmaxf(sqrtf(ss), 1e-12f);
  #pragma unroll
  for (int j = 0; j < 8; ++j) {
    const int r = t + j * 32;
    qb[((size_t)(l * R_ + r)) * B_ + b] = f2bf(v[j] * inv);
  }
}

// ---------------------------------------------------------------------------
// Kernel 3: scores[b,d] = mean_l max_c dot(doc_keys[d,l,c,:]/||.||, qhat[b,l,:])
// 256 blocks x 8 waves; one doc per wave; per-wave private LDS double buffer.
// Stage fill alternates between the two CDNA5 DMA paths:
//   even stages: TENSOR_LOAD_TO_LDS (TDM descriptor, hw row padding, TENSORcnt)
//   odd  stages: GLOBAL_LOAD_ASYNC_TO_LDS_B128 (ASYNCcnt)
// TENSOR and ASYNC ops are in-order within their own counter, so after issuing
// stage st+1 (other type) we wait stage st's counter down to 0.
// GEMM per (d,l): [C=32 x B=16] = K[c,r] * Q^T[r,b], R = 256 (bf16 WMMA).
// Row-norm folded in as post-GEMM scale.
// ---------------------------------------------------------------------------
__global__ __launch_bounds__(256) void k3_docscore(const float* __restrict__ dk,
                                                   const unsigned short* __restrict__ qb,
                                                   float* __restrict__ scores) {
  extern __shared__ __align__(16) char smem3[];
  constexpr int SR = 132;                 // f32 per staged half-row (128 + 4 pad)
  constexpr int HALF_F32 = 32 * SR;       // one half-tile: 32 rows x 128 (padded)
  constexpr int HALF_BYTES = HALF_F32 * 4;// 16896 B
  const int t = threadIdx.x, lane = t & 31, wid = t >> 5;
  const int d = blockIdx.x * 8 + wid;
  float* sbase = (float*)smem3 + (size_t)wid * 2 * HALF_F32;  // per-wave double buffer
  // LDS byte address for DMA = low 32 bits of the flat shared-aperture address
  const unsigned lds0 = (unsigned)(uintptr_t)sbase;

  // TDM D# group1 (uniform for the whole kernel):
  //  data_size=4B, pad_enable, pad_interval=128 DWORDs, pad_amount=4 DWORDs,
  //  tensor_dim0=128, tensor_dim1=32, tile 128x32, tensor_dim0_stride=256 elems
  u32x8 g1;
  g1[0] = (2u << 16) | (1u << 20) | (6u << 22) | (3u << 25);
  g1[1] = (128u << 16);      // tensor_dim0 (low 16 at bits 63:48)
  g1[2] = (32u  << 16);      // tensor_dim1 (low 16 at bits 95:80)
  g1[3] = (128u << 16);      // tile_dim0 at bits 127:112
  g1[4] = 32u;               // tile_dim1 at bits 143:128 (tile_dim2 = 0)
  g1[5] = 256u;              // tensor_dim0_stride at bits 207:160
  g1[6] = 0u;                // dim0_stride hi / dim1_stride = 0 (2D tile)
  g1[7] = 0u;

  // one pipeline stage = half of one layer's 32x256 chunk-key tile (32x128 f32)
  auto issueStageTDM = [&](int st) {          // single TDM op per wave
    const int l = st >> 1, kh = st & 1;
    const uint64_t ga = (uint64_t)(uintptr_t)
        (dk + ((size_t)d * L_ + l) * (size_t)(C_ * R_) + kh * 128);
    u32x4 g0;
    g0[0] = RFL(1u);                                        // count=1, 2D, no gather
    g0[1] = RFL(lds0 + (unsigned)((st & 1) * HALF_BYTES));  // lds_addr (bytes)
    g0[2] = RFL((unsigned)ga);                              // global_addr[31:0]
    g0[3] = RFL(((unsigned)(ga >> 32) & 0x01FFFFFFu) | 0x80000000u); // [56:32]|type=2
    asm volatile("tensor_load_to_lds %0, %1" :: "s"(g0), "s"(g1) : "memory");
  };
  auto issueStageAsync = [&](int st) {        // 32 lane-parallel async b128 ops
    const int l = st >> 1, kh = st & 1;
    const unsigned bufofs = (unsigned)((st & 1) * HALF_BYTES);
    const float* g0p = dk + ((size_t)d * L_ + l) * (size_t)(C_ * R_) + kh * 128 + lane * 4;
    #pragma unroll
    for (int j = 0; j < 32; ++j) {            // row j: 32 lanes x 16B = full 512B row
      const unsigned laddr = lds0 + bufofs + (unsigned)(j * SR * 4) + lane * 16;
      const float* g = g0p + (size_t)j * R_;
      asm volatile("global_load_async_to_lds_b128 %0, %1, off"
                   :: "v"(laddr), "v"(g) : "memory");
    }
  };

  const v8f z = {0.f,0.f,0.f,0.f,0.f,0.f,0.f,0.f};
  float scoreacc = 0.f;
  v8f acc[2]; acc[0] = z; acc[1] = z;
  float sumsq = 0.f;

  issueStageTDM(0);
  const int NSTG = 2 * L_;  // 36 stages
  for (int st = 0; st < NSTG; ++st) {
    const int l = st >> 1, kh = st & 1;
    if (st + 1 < NSTG) {
      // all prior LDS reads of the buffer we are about to overwrite are done
      asm volatile("s_wait_dscnt 0x0" ::: "memory");
      if ((st + 1) & 1) issueStageAsync(st + 1);
      else              issueStageTDM(st + 1);
    }
    // wait for the counter that tracks THIS stage's fill (in-order per type)
    if (kh == 0) { WAIT_TENSOR(0); } else { WAIT_ASYNC(0); }
    float* s = sbase + (st & 1) * HALF_F32;

    // lane owns chunk-row 'lane': accumulate sum of squares for this half
    {
      const float4* rp = (const float4*)(s + lane * SR);
      float ssl = 0.f;
      #pragma unroll
      for (int j = 0; j < 32; ++j) {
        float4 vv = rp[j];
        ssl += vv.x * vv.x + vv.y * vv.y + vv.z * vv.z + vv.w * vv.w;
      }
      sumsq += ssl;
    }
    #pragma unroll
    for (int kt = 0; kt < 4; ++kt) {
      const int kg = kh * 128 + kt * 32;
      // B fragment from global qb: lane = column b (lane&15), k in table order
      const unsigned short* bp = qb + ((size_t)(l * R_ + kg + (lane >> 4) * 8)) * B_ + (lane & 15);
      v16bf bfr;
      #pragma unroll
      for (int e = 0; e < 16; ++e)
        bfr[e] = __builtin_bit_cast(__bf16, bp[(size_t)((e & 7) + ((e >> 3) << 4)) * B_]);
      #pragma unroll
      for (int ti = 0; ti < 2; ++ti) {   // two 16-row tiles cover C=32
        const float* ap = s + (ti * 16 + (lane & 15)) * SR + kt * 32 + (lane >> 4) * 8;
        float4 a0 = *(const float4*)ap;
        float4 a1 = *(const float4*)(ap + 4);
        float4 a2 = *(const float4*)(ap + 16);
        float4 a3 = *(const float4*)(ap + 20);
        U8Frag u;
        u.w[0] = pk_bf16x2(a0.x, a0.y); u.w[1] = pk_bf16x2(a0.z, a0.w);
        u.w[2] = pk_bf16x2(a1.x, a1.y); u.w[3] = pk_bf16x2(a1.z, a1.w);
        u.w[4] = pk_bf16x2(a2.x, a2.y); u.w[5] = pk_bf16x2(a2.z, a2.w);
        u.w[6] = pk_bf16x2(a3.x, a3.y); u.w[7] = pk_bf16x2(a3.z, a3.w);
        v16bf af = __builtin_bit_cast(v16bf, u);
        acc[ti] = __builtin_amdgcn_wmma_f32_16x16x32_bf16(
            false, af, false, bfr, (short)0, acc[ti], false, false);
      }
    }

    if (kh == 1) {
      // scale rows by 1/||row||, max over chunks, accumulate mean over layers
      const float inv = 1.f / fmaxf(sqrtf(sumsq), 1e-12f);  // for row 'lane'
      float mv = -3.402823466e38f;
      #pragma unroll
      for (int ti = 0; ti < 2; ++ti)
        #pragma unroll
        for (int r = 0; r < 8; ++r) {
          const int c = ti * 16 + ((lane >> 4) << 3) + r;   // row of this acc elem
          const float ic = __shfl(inv, c, 32);
          mv = fmaxf(mv, acc[ti][r] * ic);
        }
      mv = fmaxf(mv, __shfl_xor(mv, 16, 32));  // combine row halves; b = lane&15
      scoreacc += mv * (1.0f / L_);
      acc[0] = z; acc[1] = z; sumsq = 0.f;
    }
  }
  if (lane < 16) scores[(size_t)lane * D_ + d] = scoreacc;  // scores[b][d]
}

// ---------------------------------------------------------------------------
// Kernel 4: per-batch top-16 (descending, ties -> lower index like lax.top_k)
// out[0:256] = scores[b,j]; out[256:512] = (float)indices[b,j]
// ---------------------------------------------------------------------------
__global__ __launch_bounds__(256) void k4_topk(const float* __restrict__ scores,
                                               float* __restrict__ out) {
  __shared__ float sv[256];
  __shared__ int   si[256];
  __shared__ int   swin;
  const int t = threadIdx.x, b = blockIdx.x;
  float v[8]; int idx[8];
  #pragma unroll
  for (int j = 0; j < 8; ++j) {
    idx[j] = t + j * 256;
    v[j] = scores[(size_t)b * D_ + idx[j]];
  }
  for (int sel = 0; sel < TOPK_; ++sel) {
    float bv = -3.402823466e38f; int bi = 0x7FFFFFFF;
    #pragma unroll
    for (int j = 0; j < 8; ++j)
      if (v[j] > bv || (v[j] == bv && idx[j] < bi)) { bv = v[j]; bi = idx[j]; }
    sv[t] = bv; si[t] = bi;
    __syncthreads();
    for (int s2 = 128; s2 > 0; s2 >>= 1) {
      if (t < s2) {
        const float ov = sv[t + s2]; const int oi = si[t + s2];
        if (ov > sv[t] || (ov == sv[t] && oi < si[t])) { sv[t] = ov; si[t] = oi; }
      }
      __syncthreads();
    }
    if (t == 0) {
      out[b * TOPK_ + sel]       = sv[0];
      out[256 + b * TOPK_ + sel] = (float)si[0];
      swin = si[0];
    }
    __syncthreads();
    #pragma unroll
    for (int j = 0; j < 8; ++j)
      if (idx[j] == swin) v[j] = -3.402823466e38f;
    __syncthreads();
  }
}

// ---------------------------------------------------------------------------
extern "C" void kernel_launch(void* const* d_in, const int* in_sizes, int n_in,
                              void* d_out, int out_size, void* d_ws, size_t ws_size,
                              hipStream_t stream) {
  (void)in_sizes; (void)n_in; (void)out_size; (void)ws_size;
  const float* qhs = (const float*)d_in[0];   // [16,18,256,2560] f32
  const float* wq  = (const float*)d_in[1];   // [18,256,2560]    f32
  const float* dk  = (const float*)d_in[2];   // [2048,18,32,256] f32

  char* ws = (char*)d_ws;
  unsigned*       qkey   = (unsigned*)ws;                       // 288*256*4   = 294912 B
  unsigned short* qb     = (unsigned short*)(ws + 294912);      // 18*256*16*2 = 147456 B
  float*          scores = (float*)(ws + 294912 + 147456);      // 16*2048*4   = 131072 B

  k0_init<<<dim3(B_ * L_), 256, 0, stream>>>(qkey);

  dim3 g1(2, 2, B_ * L_);
  size_t lds1 = (size_t)(2 * 128 * 80 + 2 * 64 * 144) * 2;      // 77824 B
  k1_project<<<g1, 256, lds1, stream>>>(qhs, wq, qkey);

  k2_norm<<<dim3(B_ * L_), 32, 0, stream>>>(qkey, qb);

  size_t lds3 = (size_t)8 * 2 * 32 * 132 * 4;                   // 270336 B
  k3_docscore<<<dim3(D_ / 8), 256, lds3, stream>>>(dk, qb, scores);

  k4_topk<<<dim3(B_), 256, 0, stream>>>(scores, (float*)d_out);
}
